// VectorQuantizedSelfAttention_81389630259537
// MI455X (gfx1250) — matrix operand order repaired and checked by reading, not compiled, not run on hardware
//
#include <hip/hip_runtime.h>
#include <hip/hip_fp16.h>
#include <math.h>

// ---------------------------------------------------------------------------
// VQ self-attention forward, MI455X (gfx1250, wave32, WMMA).
// All contraction stages use v_wmma_f32_16x16x32_f16 (f16 A/B, f32 acc).
// All WMMA operands are stored K-contiguous so fragment loads are wide
// (b128) vector loads instead of strided u16 gathers.
// ---------------------------------------------------------------------------

typedef __attribute__((ext_vector_type(16))) _Float16 v16h;
typedef __attribute__((ext_vector_type(8)))  _Float16 v8h;
typedef __attribute__((ext_vector_type(8)))  float    v8f;

#define DEV __device__ __forceinline__

namespace {
constexpr int B    = 8;
constexpr int T    = 1024;
constexpr int D    = 1024;
constexpr int H    = 16;
constexpr int V    = 256;
constexpr int VAR  = 64;
constexpr int HV   = H * V;          // 4096
constexpr int NROWS = B * T * H;     // 131072  (b,t,h) rows
}

// --------------------------- WMMA fragment helpers -------------------------
// A fragment: 16x32 f16 tile, K-contiguous storage (elem (m,k) = p00[m*rs+k]).
// lane L: M = L&15; lanes 0-15 hold K 0-7,16-23; lanes 16-31 hold K 8-15,24-31
// (ISA 7.12.2 16-bit A layout). Two aligned 16B loads per lane.
DEV v16h frag_a(const _Float16* p00, long rs) {
  int l = threadIdx.x;               // blockDim.x == 32 in WMMA kernels
  const _Float16* p = p00 + (long)(l & 15) * rs + ((l >> 4) * 8);
  v8h lo = *(const v8h*)p;           // K = kb .. kb+7
  v8h hi = *(const v8h*)(p + 16);    // K = 16+kb .. 16+kb+7
  return __builtin_shufflevector(lo, hi, 0, 1, 2, 3, 4, 5, 6, 7,
                                 8, 9, 10, 11, 12, 13, 14, 15);
}

// B fragment: 32x16 f16 tile, K-contiguous storage (elem (k,n) = p00[n*ns+k]).
// lane L: N = L&15; lanes 0-15 hold K 0-7,16-23; lanes 16-31 hold K 8-15,24-31.
// Two aligned 16B loads per lane.
DEV v16h frag_b(const _Float16* p00, long ns) {
  int l = threadIdx.x;
  const _Float16* p = p00 + (long)(l & 15) * ns + ((l >> 4) * 8);
  v8h lo = *(const v8h*)p;
  v8h hi = *(const v8h*)(p + 16);
  return __builtin_shufflevector(lo, hi, 0, 1, 2, 3, 4, 5, 6, 7,
                                 8, 9, 10, 11, 12, 13, 14, 15);
}

DEV v8f wmma_f16(v16h a, v16h b, v8f c) {
  return __builtin_amdgcn_wmma_f32_16x16x32_f16(
      /*neg_a=*/false, a, /*neg_b=*/false, b,
      /*c_mod=*/(short)0, c, /*reuse_a=*/false, /*reuse_b=*/false);
}

// ------------------------------- utility kernels ---------------------------
__global__ void zero_kernel(float* cnt, float* acc) {
  int i = blockIdx.x * 256 + threadIdx.x;
  if (i < H * V) cnt[i] = 0.0f;
  if (i < 8)     acc[i] = 0.0f;
}

__global__ void cvt_f16_kernel(const float* __restrict__ src,
                               _Float16* __restrict__ dst, long n) {
  long i = (long)blockIdx.x * blockDim.x + threadIdx.x;
  if (i < n) dst[i] = (_Float16)src[i];
}

// x (B,T,D) f32 -> xhT (B,D,T) f16  (coalesced writes, strided reads)
__global__ void xpose_x_kernel(const float* __restrict__ x,
                               _Float16* __restrict__ xhT) {
  long gid = (long)blockIdx.x * blockDim.x + threadIdx.x;   // < B*D*T
  int  t  = (int)(gid & (T - 1));
  long nd = gid >> 10;               // b*D + d
  int  d  = (int)(nd & (D - 1));
  long b  = nd >> 10;
  xhT[gid] = (_Float16)x[(b * T + t) * D + d];
}

// ----------------------------- grouped 1x1 conv ----------------------------
// ze[b,t,h,o] = sum_i x[b,t,h*VAR+i] * conv_w[h,o,i]
__global__ void conv_kernel(const float* __restrict__ x,
                            const float* __restrict__ cw,
                            float* __restrict__ ze) {
  long gid = (long)blockIdx.x * blockDim.x + threadIdx.x;   // < NROWS*VAR
  int  o   = (int)(gid & 63);
  long n   = gid >> 6;               // (b,t,h)
  int  h   = (int)(n & 15);
  long bt  = n >> 4;
  const float* xp = x + bt * D + h * VAR;
  const float* wp = cw + ((long)h * VAR + o) * VAR;
  float acc = 0.0f;
#pragma unroll 8
  for (int i = 0; i < VAR; ++i) acc += xp[i] * wp[i];
  ze[gid] = acc;
}

// ------------------------ GroupNorm (per b,h over t,i) ---------------------
__global__ void gn_stats_kernel(const float* __restrict__ ze,
                                float* __restrict__ stat) {
  __shared__ float s1[256], s2[256];
  int bh = blockIdx.x;               // B*H groups
  int b = bh >> 4, h = bh & 15;
  float s = 0.0f, ss = 0.0f;
  for (int idx = threadIdx.x; idx < T * VAR; idx += 256) {
    int t = idx >> 6, i = idx & 63;
    float v = ze[((long)(b * T + t) * H + h) * VAR + i];
    s += v; ss += v * v;
  }
  s1[threadIdx.x] = s; s2[threadIdx.x] = ss;
  __syncthreads();
  for (int o = 128; o > 0; o >>= 1) {
    if (threadIdx.x < o) {
      s1[threadIdx.x] += s1[threadIdx.x + o];
      s2[threadIdx.x] += s2[threadIdx.x + o];
    }
    __syncthreads();
  }
  if (threadIdx.x == 0) {
    float inv = 1.0f / (float)(T * VAR);
    float mu  = s1[0] * inv;
    float var = s2[0] * inv - mu * mu;
    stat[bh * 2 + 0] = mu;
    stat[bh * 2 + 1] = rsqrtf(var + 1e-5f);
  }
}

__global__ void gn_apply_kernel(const float* __restrict__ stat,
                                const float* __restrict__ gnw,
                                const float* __restrict__ gnb,
                                float* __restrict__ ze,
                                _Float16* __restrict__ zeh) {
  long gid = (long)blockIdx.x * blockDim.x + threadIdx.x;   // < NROWS*VAR
  int  i   = (int)(gid & 63);
  long n   = gid >> 6;
  int  h   = (int)(n & 15);
  int  b   = (int)(n >> 14);         // n = (b*1024 + t)*16 + h
  int  bh  = b * H + h;
  float mu = stat[bh * 2 + 0];
  float rs = stat[bh * 2 + 1];
  float v  = (ze[gid] - mu) * rs * gnw[h * VAR + i] + gnb[h * VAR + i];
  ze[gid]  = v;
  zeh[gid] = (_Float16)v;
}

// row ||ze||^2 per (b,t,h); codebook ||emb_v||^2
__global__ void rowsq_kernel(const float* __restrict__ ze,
                             float* __restrict__ zsq) {
  long n = (long)blockIdx.x * blockDim.x + threadIdx.x;     // < NROWS
  const float* p = ze + n * VAR;
  float s = 0.0f;
#pragma unroll 8
  for (int i = 0; i < VAR; ++i) s += p[i] * p[i];
  zsq[n] = s;
}

__global__ void esq_kernel(const float* __restrict__ emb,
                           float* __restrict__ esq) {
  int v = threadIdx.x;               // 256 threads
  const float* p = emb + (long)v * VAR;
  float s = 0.0f;
#pragma unroll 8
  for (int i = 0; i < VAR; ++i) s += p[i] * p[i];
  esq[v] = s;
}

// --------------------- dist = sqrt(|ze|^2+|e|^2-2 ze.e) --------------------
// rows: (b,t,h) 131072; cols: V=256; K=VAR=64. WMMA.
__global__ void dist_wmma_kernel(const _Float16* __restrict__ zeh,
                                 const _Float16* __restrict__ embh,
                                 const float* __restrict__ zsq,
                                 const float* __restrict__ esq,
                                 float* __restrict__ dist) {
  int row0 = (blockIdx.x * blockDim.y + threadIdx.y) * 16;
  int col0 = blockIdx.y * 16;
  v8f acc = {};
#pragma unroll
  for (int k0 = 0; k0 < VAR; k0 += 32) {
    v16h a = frag_a(zeh + (long)row0 * VAR + k0, VAR);
    v16h b = frag_b(embh + (long)col0 * VAR + k0, VAR); // B[k][v]=emb[v][k]
    acc = wmma_f16(a, b, acc);
  }
  int l = threadIdx.x, n = l & 15, mh = (l >> 4) * 8;
#pragma unroll
  for (int r = 0; r < 8; ++r) {
    int M = row0 + mh + r, N = col0 + n;
    float d2 = zsq[M] + esq[N] - 2.0f * acc[r];
    dist[(long)M * V + N] = sqrtf(fmaxf(d2, 0.0f));
  }
}

// ------------------- argmin + kmeans loss + code histogram -----------------
__global__ void argmin_loss_kernel(const float* __restrict__ dist,
                                   const float* __restrict__ ze,
                                   const float* __restrict__ emb,
                                   float* __restrict__ tgt_out,
                                   float* __restrict__ cnt,
                                   float* __restrict__ acc) {
  __shared__ float red[256];
  long n = (long)blockIdx.x * 256 + threadIdx.x;            // < NROWS
  const float* row = dist + n * V;
  float best = row[0]; int bi = 0;
  for (int v = 1; v < V; ++v) {
    float d = row[v];
    if (d < best) { best = d; bi = v; }
  }
  tgt_out[n] = (float)bi;
  int h = (int)(n & 15);
  atomicAdd(&cnt[h * V + bi], 1.0f);
  const float* e = emb + (long)bi * VAR;
  const float* z = ze + n * VAR;
  float sq = 0.0f;
#pragma unroll 8
  for (int i = 0; i < VAR; ++i) { float df = e[i] - z[i]; sq += df * df; }
  red[threadIdx.x] = sq;
  __syncthreads();
  for (int o = 128; o > 0; o >>= 1) {
    if (threadIdx.x < o) red[threadIdx.x] += red[threadIdx.x + o];
    __syncthreads();
  }
  if (threadIdx.x == 0) atomicAdd(acc, red[0]);
}

// --------------------------- softmaxes (VALU) ------------------------------
// prob over time (masked); OUTPUT TRANSPOSED to (B,H,V,T) f16 so the e_new
// GEMM's A operand is K(t)-contiguous. Writes are fully coalesced per thread.
__global__ void softmax_time_kernel(const float* __restrict__ dist,
                                    const unsigned char* __restrict__ mask,
                                    const float* __restrict__ temp,
                                    _Float16* __restrict__ pthT) {
  int tid = blockIdx.x * blockDim.x + threadIdx.x;          // < B*H*V
  int b = tid / HV; int rem = tid - b * HV;
  int h = rem >> 8; int v = rem & 255;
  float scale = expf(temp[0]);
  long base  = (long)b * T * HV + (long)h * V + v;          // + t*HV (read)
  long obase = (long)tid * T;                               // + t     (write)
  float mx = -INFINITY;
  for (int t = 0; t < T; ++t) {
    float lg = mask[b * T + t] ? -INFINITY : -dist[base + (long)t * HV] * scale;
    mx = fmaxf(mx, lg);
  }
  float sum = 0.0f;
  for (int t = 0; t < T; ++t) {
    float lg = mask[b * T + t] ? -INFINITY : -dist[base + (long)t * HV] * scale;
    sum += (lg == -INFINITY) ? 0.0f : expf(lg - mx);
  }
  float inv = (sum > 0.0f) ? 1.0f / sum : 0.0f;
  for (int t = 0; t < T; ++t) {
    float lg = mask[b * T + t] ? -INFINITY : -dist[base + (long)t * HV] * scale;
    float e  = (lg == -INFINITY) ? 0.0f : expf(lg - mx);
    pthT[obase + t] = (_Float16)(e * inv);
  }
}

// prob over codes (unmasked), contiguous row of 256 -> f16
__global__ void softmax_v_kernel(const float* __restrict__ dist,
                                 const float* __restrict__ temp,
                                 _Float16* __restrict__ pvh) {
  long n = (long)blockIdx.x * blockDim.x + threadIdx.x;     // < NROWS
  const float* row = dist + n * V;
  float scale = expf(temp[0]);
  float mx = -INFINITY;
  for (int v = 0; v < V; ++v) mx = fmaxf(mx, -row[v] * scale);
  float sum = 0.0f;
  for (int v = 0; v < V; ++v) sum += expf(-row[v] * scale - mx);
  float inv = 1.0f / sum;
  for (int v = 0; v < V; ++v)
    pvh[n * V + v] = (_Float16)(expf(-row[v] * scale - mx) * inv);
}

// attn = softmax(score) in place on d_out; also emit f16 copy for out-GEMM
__global__ void attn_softmax_kernel(float* __restrict__ attn,
                                    _Float16* __restrict__ atth) {
  long n = (long)blockIdx.x * blockDim.x + threadIdx.x;     // < NROWS
  float* row = attn + n * V;
  float mx = -INFINITY;
  for (int v = 0; v < V; ++v) mx = fmaxf(mx, row[v]);
  float sum = 0.0f;
  for (int v = 0; v < V; ++v) sum += expf(row[v] - mx);
  float inv = 1.0f / sum;
  for (int v = 0; v < V; ++v) {
    float p = expf(row[v] - mx) * inv;
    row[v] = p;
    atth[n * V + v] = (_Float16)p;
  }
}

// ------------------ e_new[b,h] = prob_t^T (VxT) @ x (TxD) ------------------
// Dominant GEMM (~69 GFLOP). Both operands K-contiguous; 4x unrolled K loop
// plus prefetch of the next K tile.
__global__ void enew_wmma_kernel(const _Float16* __restrict__ pthT,
                                 const _Float16* __restrict__ xhT,
                                 _Float16* __restrict__ enh) {
  int col0 = (blockIdx.x * blockDim.y + threadIdx.y) * 16;  // d
  int row0 = blockIdx.y * 16;                               // v
  int bh = blockIdx.z; int b = bh >> 4;
  // A(m=v, k=t) = pthT[(bh*V + v)*T + t]
  const _Float16* pA = pthT + ((long)bh * V + row0) * T;
  // B(k=t, n=d) = xhT[(b*D + d)*T + t]
  const _Float16* pB = xhT + ((long)b * D + col0) * T;
  v8f acc = {};
#pragma unroll 4
  for (int k0 = 0; k0 < T; k0 += 32) {
    __builtin_prefetch(pA + k0 + 128, 0, 1);
    __builtin_prefetch(pB + k0 + 128, 0, 1);
    v16h a  = frag_a(pA + k0, T);
    v16h b2 = frag_b(pB + k0, T);
    acc = wmma_f16(a, b2, acc);
  }
  int l = threadIdx.x, n = l & 15, mh = (l >> 4) * 8;
#pragma unroll
  for (int r = 0; r < 8; ++r) {
    int M = row0 + mh + r, N = col0 + n;
    enh[((long)bh * V + M) * D + N] = (_Float16)acc[r];
  }
}

// -------------------- value = e_new (BHVxD) @ wv^T (DxVAR) -----------------
// Output stored TRANSPOSED as valhT (BH,VAR,V) so the out-GEMM B operand is
// K(v)-contiguous.
__global__ void value_wmma_kernel(const _Float16* __restrict__ enh,
                                  const _Float16* __restrict__ wvh,
                                  const float* __restrict__ bv,
                                  _Float16* __restrict__ valhT) {
  int row0 = (blockIdx.x * blockDim.y + threadIdx.y) * 16;  // bhv rows
  int col0 = blockIdx.y * 16;                               // o
  v8f acc = {};
#pragma unroll 4
  for (int k0 = 0; k0 < D; k0 += 32) {
    v16h a  = frag_a(enh + (long)row0 * D + k0, D);
    v16h b2 = frag_b(wvh + (long)col0 * D + k0, D);   // B[k][o]=wv[o][k]
    acc = wmma_f16(a, b2, acc);
  }
  int l = threadIdx.x, n = l & 15, mh = (l >> 4) * 8;
#pragma unroll
  for (int r = 0; r < 8; ++r) {
    int M = row0 + mh + r, N = col0 + n;              // M = bh*V + v
    int bh = M >> 8, vv = M & 255;
    valhT[((long)bh * VAR + N) * V + vv] = (_Float16)(acc[r] + bv[N]);
  }
}

// -------------------------- q/k projections + sim --------------------------
__global__ void qk_kernel(const float* __restrict__ emb,
                          const float* __restrict__ wq, const float* __restrict__ bq,
                          const float* __restrict__ wk, const float* __restrict__ bk,
                          float* __restrict__ q, float* __restrict__ k) {
  int tid = blockIdx.x * blockDim.x + threadIdx.x;          // < V*VAR
  int v = tid >> 6, o = tid & 63;
  const float* e = emb + (long)v * VAR;
  float sq = 0.0f, sk = 0.0f;
#pragma unroll 8
  for (int i = 0; i < VAR; ++i) {
    sq += e[i] * wq[o * VAR + i];
    sk += e[i] * wk[o * VAR + i];
  }
  q[tid] = sq + bq[o];
  k[tid] = sk + bk[o];
}

// sim to d_out (row-major) + f16 copy stored COLUMN-major (simhT[w*V+v]) so
// the score-GEMM B operand is K(v)-contiguous.
__global__ void sim_kernel(const float* __restrict__ q,
                           const float* __restrict__ k,
                           float* __restrict__ sim_out,
                           _Float16* __restrict__ simhT) {
  int tid = blockIdx.x * blockDim.x + threadIdx.x;          // < V*V
  int v = tid >> 8, w = tid & 255;
  float s = 0.0f;
#pragma unroll 8
  for (int o = 0; o < VAR; ++o) s += q[v * VAR + o] * k[w * VAR + o];
  s *= 0.125f;                       // 1/sqrt(VAR)
  sim_out[tid] = s;
  simhT[(long)w * V + v] = (_Float16)s;
}

// ---------------------- score = prob_v (Nx256) @ sim -----------------------
__global__ void score_wmma_kernel(const _Float16* __restrict__ pvh,
                                  const _Float16* __restrict__ simhT,
                                  float* __restrict__ score) {
  int row0 = (blockIdx.x * blockDim.y + threadIdx.y) * 16;
  int col0 = blockIdx.y * 16;
  v8f acc = {};
#pragma unroll 2
  for (int k0 = 0; k0 < V; k0 += 32) {
    v16h a  = frag_a(pvh + (long)row0 * V + k0, V);
    v16h b2 = frag_b(simhT + (long)col0 * V + k0, V); // B[k][w]=simhT[w*V+k]
    acc = wmma_f16(a, b2, acc);
  }
  int l = threadIdx.x, n = l & 15, mh = (l >> 4) * 8;
#pragma unroll
  for (int r = 0; r < 8; ++r) {
    int M = row0 + mh + r, N = col0 + n;
    score[(long)M * V + N] = acc[r];
  }
}

// -------------- out[b,t,h,:] = attn[b,t,h,:] @ value[b,h] (256xVAR) --------
// M dim tiles over t (B operand fixed per (b,h)); both operands K-contiguous.
__global__ void out_wmma_kernel(const _Float16* __restrict__ atth,
                                const _Float16* __restrict__ valhT,
                                float* __restrict__ out) {
  int row0 = blockIdx.x * 16;        // t
  int col0 = threadIdx.y * 16;       // c within VAR
  int bh = blockIdx.z; int b = bh >> 4, h = bh & 15;
  // A(m=t, k=v) = atth[b*T*HV + t*HV + h*V + v]   (k contiguous)
  const _Float16* pA = atth + (long)b * T * HV + (long)row0 * HV + (long)h * V;
  // B(k=v, n=c) = valhT[(bh*VAR + c)*V + v]       (k contiguous)
  const _Float16* pB = valhT + (long)bh * VAR * V + (long)col0 * V;
  v8f acc = {};
#pragma unroll 2
  for (int k0 = 0; k0 < V; k0 += 32) {
    v16h a  = frag_a(pA + k0, HV);
    v16h b2 = frag_b(pB + k0, V);
    acc = wmma_f16(a, b2, acc);
  }
  int l = threadIdx.x, n = l & 15, mh = (l >> 4) * 8;
#pragma unroll
  for (int r = 0; r < 8; ++r) {
    int Mt = row0 + mh + r, N = col0 + n;
    out[((long)b * T + Mt) * D + h * VAR + N] = acc[r];
  }
}

// ------------------------------ finalize -----------------------------------
__global__ void finalize_kernel(const float* __restrict__ cnt,
                                const float* __restrict__ acc,
                                float* __restrict__ loss_out,
                                float* __restrict__ ppl_out) {
  __shared__ float ppl[16];
  int h = threadIdx.x;
  if (h < 16) {
    float s = 0.0f;
    for (int v = 0; v < V; ++v) {
      float p = cnt[h * V + v] / 8192.0f;
      s += p * logf(p + 1e-7f);
    }
    ppl[h] = expf(-s);
  }
  __syncthreads();
  if (threadIdx.x == 0) {
    float t = 0.0f;
    for (int i = 0; i < 16; ++i) t += ppl[i];
    *ppl_out = t;
    // latent == commit in forward value -> (1 + gamma) * mean
    *loss_out = acc[0] * 1.25f / (float)((long)NROWS * VAR);
  }
}

// ------------------------------ host launcher ------------------------------
extern "C" void kernel_launch(void* const* d_in, const int* in_sizes, int n_in,
                              void* d_out, int out_size, void* d_ws, size_t ws_size,
                              hipStream_t stream) {
  (void)in_sizes; (void)n_in; (void)out_size; (void)ws_size;

  const float*         x      = (const float*)d_in[0];
  const unsigned char* mask   = (const unsigned char*)d_in[1];
  const float*         conv_w = (const float*)d_in[2];
  const float*         gn_w   = (const float*)d_in[3];
  const float*         gn_b   = (const float*)d_in[4];
  const float*         emb    = (const float*)d_in[5];
  const float*         wq     = (const float*)d_in[6];
  const float*         bq     = (const float*)d_in[7];
  const float*         wk     = (const float*)d_in[8];
  const float*         bk     = (const float*)d_in[9];
  const float*         wv     = (const float*)d_in[10];
  const float*         bv     = (const float*)d_in[11];
  const float*         temp   = (const float*)d_in[12];

  // Output layout (flat, return order): out, attn, kmeans_loss, code_ppl,
  // targets (as float), sim.
  float* out_p  = (float*)d_out;
  float* attn_p = out_p + (long)B * T * D;          // NROWS * V
  float* loss_p = attn_p + (long)NROWS * V;
  float* ppl_p  = loss_p + 1;
  float* tgt_p  = ppl_p + 1;                        // NROWS
  float* sim_p  = tgt_p + NROWS;                    // V*V

  // Workspace carve-up (all 256B aligned -> vector loads stay aligned)
  char* w = (char*)d_ws;
  size_t off = 0;
  auto take = [&](size_t bytes) {
    void* p = w + off;
    off = (off + bytes + 255) & ~(size_t)255;
    return p;
  };
  float*     ze    = (float*)take((size_t)NROWS * VAR * 4);
  _Float16*  zeh   = (_Float16*)take((size_t)NROWS * VAR * 2);
  float*     zsq   = (float*)take((size_t)NROWS * 4);
  _Float16*  embh  = (_Float16*)take((size_t)V * VAR * 2);
  float*     esq   = (float*)take((size_t)V * 4);
  float*     dist  = (float*)take((size_t)NROWS * V * 4);
  _Float16*  pthT  = (_Float16*)take((size_t)NROWS * V * 2);  // (B,H,V,T)
  _Float16*  pvh   = (_Float16*)take((size_t)NROWS * V * 2);  // (N,V)
  _Float16*  xhT   = (_Float16*)take((size_t)B * D * T * 2);  // (B,D,T)
  _Float16*  enh   = (_Float16*)take((size_t)B * H * V * D * 2);
  _Float16*  wvh   = (_Float16*)take((size_t)VAR * D * 2);
  _Float16*  valhT = (_Float16*)take((size_t)B * H * VAR * V * 2); // (BH,VAR,V)
  _Float16*  simhT = (_Float16*)take((size_t)V * V * 2);      // col-major
  float*     qb    = (float*)take((size_t)V * VAR * 4);
  float*     kb    = (float*)take((size_t)V * VAR * 4);
  _Float16*  atth  = (_Float16*)take((size_t)NROWS * V * 2);
  float*     cnt   = (float*)take((size_t)H * V * 4);
  float*     stat  = (float*)take((size_t)B * H * 2 * 4);
  float*     acc   = (float*)take((size_t)8 * 4);

  const dim3 wblk(32, 4);   // 4 waves / block, one 16x16 tile per wave

  // 0) zero accumulators (graph replay safety: ws is never re-poisoned)
  zero_kernel<<<16, 256, 0, stream>>>(cnt, acc);

  // 1) grouped conv + group norm
  conv_kernel<<<(NROWS * VAR) / 256, 256, 0, stream>>>(x, conv_w, ze);
  gn_stats_kernel<<<B * H, 256, 0, stream>>>(ze, stat);
  gn_apply_kernel<<<(NROWS * VAR) / 256, 256, 0, stream>>>(stat, gn_w, gn_b, ze, zeh);

  // 2) f16 operand prep (x transposed to (B,D,T) for K-contiguous B loads)
  xpose_x_kernel<<<(B * D * T) / 256, 256, 0, stream>>>(x, xhT);
  cvt_f16_kernel<<<(V * VAR) / 256, 256, 0, stream>>>(emb, embh, (long)V * VAR);
  cvt_f16_kernel<<<(VAR * D) / 256, 256, 0, stream>>>(wv, wvh, (long)VAR * D);
  esq_kernel<<<1, 256, 0, stream>>>(emb, esq);
  rowsq_kernel<<<NROWS / 256, 256, 0, stream>>>(ze, zsq);

  // 3) distances via WMMA  (131072x256, K=64)
  dist_wmma_kernel<<<dim3(NROWS / 16 / 4, V / 16), wblk, 0, stream>>>(
      zeh, embh, zsq, esq, dist);

  // 4) argmin / kmeans loss / code histogram
  argmin_loss_kernel<<<NROWS / 256, 256, 0, stream>>>(dist, ze, emb, tgt_p, cnt, acc);

  // 5) softmaxes
  softmax_time_kernel<<<(B * H * V) / 256, 256, 0, stream>>>(dist, mask, temp, pthT);
  softmax_v_kernel<<<NROWS / 256, 256, 0, stream>>>(dist, temp, pvh);

  // 6) e_new = prob_t^T @ x per (b,h)  (256x1024, K=1024) -- dominant GEMM
  enew_wmma_kernel<<<dim3(D / 16 / 4, V / 16, B * H), wblk, 0, stream>>>(pthT, xhT, enh);

  // 7) value = e_new @ wv^T + bv  (32768x64, K=1024), transposed store
  value_wmma_kernel<<<dim3(B * H * V / 16 / 4, VAR / 16), wblk, 0, stream>>>(
      enh, wvh, bv, valhT);

  // 8) q/k/sim
  qk_kernel<<<(V * VAR) / 256, 256, 0, stream>>>(emb, wq, bq, wk, bk, qb, kb);
  sim_kernel<<<(V * V) / 256, 256, 0, stream>>>(qb, kb, sim_p, simhT);

  // 9) score = prob_v @ sim  (131072x256, K=256) -> attn region, then softmax
  score_wmma_kernel<<<dim3(NROWS / 16 / 4, V / 16), wblk, 0, stream>>>(
      pvh, simhT, attn_p);
  attn_softmax_kernel<<<NROWS / 256, 256, 0, stream>>>(attn_p, atth);

  // 10) out = attn @ value  (per (b,h): 1024x64, K=256)
  out_wmma_kernel<<<dim3(T / 16, 1, B * H), wblk, 0, stream>>>(atth, valhT, out_p);

  // 11) scalar outputs
  finalize_kernel<<<1, 32, 0, stream>>>(cnt, acc, loss_p, ppl_p);
}
